// IPSingleStreamBlockProcessor_37117107372191
// MI455X (gfx1250) — matrix operand order repaired and checked by reading, not compiled, not run on hardware
//
#include <hip/hip_runtime.h>
#include <hip/hip_bf16.h>
#include <math.h>

// ---------------------------------------------------------------------------
// Types for CDNA5 WMMA (gfx1250, wave32)
// ---------------------------------------------------------------------------
typedef __attribute__((ext_vector_type(16))) __bf16          v16bf;
typedef __attribute__((ext_vector_type(8)))  float           v8f;
typedef __attribute__((ext_vector_type(8)))  unsigned short  ushort8;
typedef __attribute__((ext_vector_type(4)))  unsigned int    u32x4;
typedef __attribute__((ext_vector_type(8)))  int             i32x8;
typedef __attribute__((ext_vector_type(4)))  int             i32x4;

#define HID_  3072
#define Hh    24
#define Dd    128
#define Ll    2048
#define LIP_  64
#define CTX_  4096
#define MLP_  12288
#define FUSED_ (HID_ + MLP_)   // 15360
#define QKV_  (3 * HID_)       // 9216

// ---------------------------------------------------------------------------
// TDM availability (Tensor Data Mover: tensor_load_to_lds)
// ---------------------------------------------------------------------------
#if defined(__has_builtin)
# if __has_builtin(__builtin_amdgcn_tensor_load_to_lds) && \
     __has_builtin(__builtin_amdgcn_s_wait_tensorcnt)
#  define HAVE_TDM 1
# endif
#endif
#ifndef HAVE_TDM
# define HAVE_TDM 0
#endif
#if defined(__clang_major__) && (__clang_major__ >= 23)
# define TDM_ARITY6 1
#else
# define TDM_ARITY6 0
#endif

// ---------------------------------------------------------------------------
// Helpers
// ---------------------------------------------------------------------------
__device__ __forceinline__ unsigned short f2bf(float f) {
  unsigned u = __builtin_bit_cast(unsigned, f);
  u += 0x7FFFu + ((u >> 16) & 1u);   // round-to-nearest-even
  return (unsigned short)(u >> 16);
}

__device__ __forceinline__ float gelu_tanh(float x) {
  float x3 = x * x * x;
  return 0.5f * x * (1.0f + tanhf(0.7978845608028654f * (x + 0.044715f * x3)));
}

__device__ __forceinline__ v8f wmma_bf16(v16bf a, v16bf b, v8f c) {
  // D(16x16 f32) = A(16x32 bf16) * B(32x16 bf16) + C
  return __builtin_amdgcn_wmma_f32_16x16x32_bf16(
      /*neg_a=*/false, a, /*neg_b=*/false, b,
      /*c_mod=*/(short)0, c, /*reuse_a=*/false, /*reuse_b=*/false);
}

__device__ __forceinline__ unsigned ldsOffset(const void* p) {
  // Generic LDS-aperture address: low 32 bits are the LDS byte offset.
  return (unsigned)(uintptr_t)p;
}

#if HAVE_TDM
// Issue a 2D tile DMA (global -> LDS) via the Tensor Data Mover.
//   data_size = 2 bytes (bf16 elements). Rows beyond tensor_h are zero-filled.
//   pad_* add LDS padding after each row to produce a banked row stride.
__device__ __forceinline__ void tdm_load_tile_2d(
    unsigned lds_byte_off, const void* gaddr,
    unsigned tensor_w, unsigned tensor_h,
    unsigned tile_w, unsigned tile_h,
    unsigned row_stride_elems,
    unsigned pad_interval_code, unsigned pad_amount_code) {
  unsigned long long ga = (unsigned long long)(uintptr_t)gaddr;
  u32x4 g0;
  g0[0] = 1u;                                        // count=1, user mode
  g0[1] = lds_byte_off;                              // lds_addr
  g0[2] = (unsigned)(ga & 0xFFFFFFFFu);              // global_addr[31:0]
  g0[3] = (unsigned)((ga >> 32) & 0x01FFFFFFu)       // global_addr[56:32]
          | (2u << 30);                              // type = 2 ("image")
  i32x8 g1;
  unsigned w0 = (1u << 16)                           // data_size = 1 -> 2 bytes
              | (1u << 20)                           // pad_enable
              | (pad_interval_code << 22)
              | (pad_amount_code << 25);
  g1[0] = (int)w0;
  g1[1] = (int)((tensor_w & 0xFFFFu) << 16);                              // dim0[15:0]
  g1[2] = (int)(((tensor_w >> 16) & 0xFFFFu) | ((tensor_h & 0xFFFFu) << 16));
  g1[3] = (int)(((tensor_h >> 16) & 0xFFFFu) | ((tile_w & 0xFFFFu) << 16));
  g1[4] = (int)(tile_h & 0xFFFFu);                                        // tile_dim1
  g1[5] = (int)row_stride_elems;                                          // dim0_stride[31:0]
  g1[6] = 0;
  g1[7] = 0;
  i32x4 z4;
  z4[0] = 0; z4[1] = 0; z4[2] = 0; z4[3] = 0;
#if TDM_ARITY6
  i32x8 z8;
  #pragma unroll
  for (int i = 0; i < 8; i++) z8[i] = 0;
  __builtin_amdgcn_tensor_load_to_lds(g0, g1, z4, z4, z8, 0);
#else
  __builtin_amdgcn_tensor_load_to_lds(g0, g1, z4, z4, 0);
#endif
}
#endif  // HAVE_TDM

// A-operand fragment (16x32, lane = M row, K interleaved {0..7,16..23}/{8..15,24..31})
__device__ __forceinline__ v16bf ldsFragA(const unsigned short* base, int stride,
                                          int row0, int k0) {
  int lane = threadIdx.x & 31;
  int r  = row0 + (lane & 15);
  int kk = k0 + ((lane >> 4) << 3);
  const unsigned short* p = base + r * stride + kk;
  union { v16bf v; ushort8 u[2]; } t;
  t.u[0] = *reinterpret_cast<const ushort8*>(p);
  t.u[1] = *reinterpret_cast<const ushort8*>(p + 16);
  return t.v;
}

// B-operand fragment (32x16 = K x N). Source rows are columns of B (i.e. W rows,
// K-contiguous). K interleaved {0..7,16..23}/{8..15,24..31} to match the A fragment.
__device__ __forceinline__ v16bf ldsFragB(const unsigned short* base, int stride,
                                          int row0, int k0) {
  int lane = threadIdx.x & 31;
  int r  = row0 + (lane & 15);
  int kk = k0 + ((lane >> 4) << 3);
  const unsigned short* p = base + r * stride + kk;
  union { v16bf v; ushort8 u[2]; } t;
  t.u[0] = *reinterpret_cast<const ushort8*>(p);
  t.u[1] = *reinterpret_cast<const ushort8*>(p + 16);
  return t.v;
}

__device__ __forceinline__ float blockReduceSum(float v, float* red) {
  #pragma unroll
  for (int m = 16; m >= 1; m >>= 1) v += __shfl_xor(v, m, 32);
  int wid = threadIdx.x >> 5, lane = threadIdx.x & 31;
  if (lane == 0) red[wid] = v;
  __syncthreads();
  float s = 0.f;
  int nw = blockDim.x >> 5;
  for (int i = 0; i < nw; i++) s += red[i];
  __syncthreads();
  return s;
}

// ---------------------------------------------------------------------------
// 1. mod = silu(vec) @ mod_lin_w.T + mod_lin_b    (9216 outputs, K=3072)
// ---------------------------------------------------------------------------
__global__ __launch_bounds__(256) void k_mod_gemv(
    const float* __restrict__ vec, const float* __restrict__ W,
    const float* __restrict__ b, float* __restrict__ mod) {
  int n    = blockIdx.x * 8 + (threadIdx.x >> 5);
  int lane = threadIdx.x & 31;
  float s = 0.f;
  for (int k = lane; k < HID_; k += 32) {
    float v  = vec[k];
    float sv = v / (1.f + __expf(-v));
    s += sv * W[(size_t)n * HID_ + k];
  }
  #pragma unroll
  for (int m = 16; m >= 1; m >>= 1) s += __shfl_xor(s, m, 32);
  if (lane == 0) mod[n] = s + b[n];
}

// ---------------------------------------------------------------------------
// 2. x_mod = (1+scale)*layernorm(x) + shift   -> bf16
// ---------------------------------------------------------------------------
__global__ __launch_bounds__(256) void k_ln_modulate(
    const float* __restrict__ x, const float* __restrict__ mod,
    unsigned short* __restrict__ xmod) {
  __shared__ float red[8];
  int l = blockIdx.x;
  const float* row = x + (size_t)l * HID_;
  float vals[12];
  float s = 0.f, ss = 0.f;
  #pragma unroll
  for (int i = 0; i < 12; i++) {
    float v = row[threadIdx.x + i * 256];
    vals[i] = v; s += v; ss += v * v;
  }
  float mean = blockReduceSum(s, red) * (1.0f / HID_);
  float var  = blockReduceSum(ss, red) * (1.0f / HID_) - mean * mean;
  float rstd = rsqrtf(var + 1e-6f);
  #pragma unroll
  for (int i = 0; i < 12; i++) {
    int c = threadIdx.x + i * 256;
    float sc = mod[HID_ + c], sh = mod[c];
    xmod[(size_t)l * HID_ + c] = f2bf((1.f + sc) * ((vals[i] - mean) * rstd) + sh);
  }
}

// ---------------------------------------------------------------------------
// 3. fp32 -> bf16 elementwise
// ---------------------------------------------------------------------------
__global__ void k_f32_to_bf16(const float* __restrict__ in,
                              unsigned short* __restrict__ out, int n) {
  int i = blockIdx.x * 256 + threadIdx.x;
  if (i < n) out[i] = f2bf(in[i]);
}

// ---------------------------------------------------------------------------
// 4. Generic WMMA GEMM:  C[M,N] = A(bf16, MxK row-major) * W^T (W fp32, NxK)
//    MODE 0: fp32 store (+bias)            MODE 1: gelu -> bf16 (+bias)
//    MODE 2: bf16 store (+optional bias)   MODE 3: out = x + gate*(C+bias)  fp32
// ---------------------------------------------------------------------------
#define BM 128
#define BN 128
#define BKK 32
#define LDT 40   // padded LDS row stride (halfs): 32 + 4 DWORDs pad

template <int MODE>
__global__ __launch_bounds__(256) void k_gemm(
    const unsigned short* __restrict__ A, int lda,
    const float* __restrict__ W, int ldw,
    const float* __restrict__ bias,
    void* __restrict__ outv, int ldc,
    int M, int N, int K,
    const float* __restrict__ xres, const float* __restrict__ gate) {
  __shared__ unsigned short As[BM * LDT];
  __shared__ unsigned short Bs[BN * LDT];
  int tid   = threadIdx.x;
  int lane  = tid & 31;
  int wave  = tid >> 5;
  int waveM = wave >> 2;     // 0..1  (64 rows each)
  int waveN = wave & 3;      // 0..3  (32 cols each)
  int m0 = blockIdx.y * BM, n0 = blockIdx.x * BN;

  v8f acc[4][2];
  #pragma unroll
  for (int mi = 0; mi < 4; mi++)
    #pragma unroll
    for (int ni = 0; ni < 2; ni++)
      #pragma unroll
      for (int j = 0; j < 8; j++) acc[mi][ni][j] = 0.f;

  int arow = tid >> 1, acol = (tid & 1) * 16;
  unsigned mRem = (unsigned)(M - m0 < BM ? M - m0 : BM);

  for (int k0 = 0; k0 < K; k0 += BKK) {
    // stage A tile (bf16) --------------------------------------------------
#if HAVE_TDM
    if (wave == 0)   // TDM: 128x32 bf16 tile, zero-fill rows >= mRem,
                     // LDS pad 4 DWORDs after every 16 DWORDs -> stride 40 halfs
      tdm_load_tile_2d(ldsOffset(As), A + (size_t)m0 * lda + k0,
                       /*tensor_w=*/BKK, /*tensor_h=*/mRem,
                       /*tile_w=*/BKK, /*tile_h=*/BM,
                       /*stride=*/(unsigned)lda,
                       /*pad_interval(16 dw)=*/3, /*pad_amount(4 dw)=*/3);
#else
    {
      ushort8 a0, a1;
      if (m0 + arow < M) {
        const ushort8* ap =
            reinterpret_cast<const ushort8*>(A + (size_t)(m0 + arow) * lda + k0 + acol);
        a0 = ap[0]; a1 = ap[1];
      } else {
        #pragma unroll
        for (int e = 0; e < 8; e++) { a0[e] = 0; a1[e] = 0; }
      }
      *reinterpret_cast<ushort8*>(As + arow * LDT + acol)     = a0;
      *reinterpret_cast<ushort8*>(As + arow * LDT + acol + 8) = a1;
    }
#endif

    // stage W tile, fp32 -> bf16 ------------------------------------------
    {
      const float4* wp =
          reinterpret_cast<const float4*>(W + (size_t)(n0 + arow) * ldw + k0 + acol);
      __align__(16) unsigned short wtmp[16];
      #pragma unroll
      for (int q = 0; q < 4; q++) {
        float4 f = wp[q];
        wtmp[q * 4 + 0] = f2bf(f.x); wtmp[q * 4 + 1] = f2bf(f.y);
        wtmp[q * 4 + 2] = f2bf(f.z); wtmp[q * 4 + 3] = f2bf(f.w);
      }
      *reinterpret_cast<ushort8*>(Bs + arow * LDT + acol)     = *reinterpret_cast<ushort8*>(&wtmp[0]);
      *reinterpret_cast<ushort8*>(Bs + arow * LDT + acol + 8) = *reinterpret_cast<ushort8*>(&wtmp[8]);
    }

    if (k0 + BKK < K)  // stream next weight tile toward L2 (global_prefetch_b8)
      __builtin_prefetch(W + (size_t)(n0 + arow) * ldw + k0 + BKK + acol, 0, 1);

#if HAVE_TDM
    if (wave == 0) __builtin_amdgcn_s_wait_tensorcnt(0);
#endif
    __syncthreads();

    v16bf af[4];
    #pragma unroll
    for (int mi = 0; mi < 4; mi++)
      af[mi] = ldsFragA(As, LDT, waveM * 64 + mi * 16, 0);
    #pragma unroll
    for (int ni = 0; ni < 2; ni++) {
      v16bf bfg = ldsFragB(Bs, LDT, waveN * 32 + ni * 16, 0);
      #pragma unroll
      for (int mi = 0; mi < 4; mi++)
        acc[mi][ni] = wmma_bf16(af[mi], bfg, acc[mi][ni]);
    }
    __syncthreads();
  }

  // epilogue -------------------------------------------------------------
  #pragma unroll
  for (int mi = 0; mi < 4; mi++)
    #pragma unroll
    for (int ni = 0; ni < 2; ni++)
      #pragma unroll
      for (int j = 0; j < 8; j++) {
        int m = m0 + waveM * 64 + mi * 16 + ((lane >> 4) << 3) + j;
        int n = n0 + waveN * 32 + ni * 16 + (lane & 15);
        if (m >= M) continue;
        float r = acc[mi][ni][j];
        if (bias) r += bias[n];
        if (MODE == 0) {
          reinterpret_cast<float*>(outv)[(size_t)m * ldc + n] = r;
        } else if (MODE == 1) {
          reinterpret_cast<unsigned short*>(outv)[(size_t)m * ldc + n] = f2bf(gelu_tanh(r));
        } else if (MODE == 2) {
          reinterpret_cast<unsigned short*>(outv)[(size_t)m * ldc + n] = f2bf(r);
        } else {
          reinterpret_cast<float*>(outv)[(size_t)m * ldc + n] =
              xres[(size_t)m * ldc + n] + gate[n] * r;
        }
      }
}

// ---------------------------------------------------------------------------
// 5. qkv postprocess: rmsnorm(q), rmsnorm(k), rope(q), rope(k) -> bf16 buffers
//    Layout of outputs: [L][H*D] halfs (head h at column h*128)
// ---------------------------------------------------------------------------
__global__ __launch_bounds__(128) void k_qkv_postproc(
    const float* __restrict__ qkv, const float* __restrict__ pe,
    const float* __restrict__ qscale, const float* __restrict__ kscale,
    unsigned short* __restrict__ qn, unsigned short* __restrict__ kn,
    unsigned short* __restrict__ vn, unsigned short* __restrict__ qnip) {
  __shared__ float red[4];
  int l = blockIdx.x;
  int d = threadIdx.x;       // 0..127
  for (int h = 0; h < Hh; h++) {
    const float* base = qkv + (size_t)l * QKV_ + h * Dd;
    float q = base[0 * HID_ + d];
    float k = base[1 * HID_ + d];
    float v = base[2 * HID_ + d];
    float msq = blockReduceSum(q * q, red) * (1.0f / Dd);
    float qh  = q * rsqrtf(msq + 1e-6f) * qscale[d];
    float msk = blockReduceSum(k * k, red) * (1.0f / Dd);
    float kh  = k * rsqrtf(msk + 1e-6f) * kscale[d];
    // rope: pe shape (L, D/2, 2, 2)
    int i = d >> 1, c = d & 1;
    const float* pp = pe + (((size_t)l * (Dd / 2) + i) * 2 + c) * 2;
    float a = pp[0], b = pp[1];
    float qpart = __shfl_xor(qh, 1, 32);
    float kpart = __shfl_xor(kh, 1, 32);
    float q0 = c ? qpart : qh, q1 = c ? qh : qpart;
    float k0 = c ? kpart : kh, k1 = c ? kh : kpart;
    size_t o = (size_t)l * HID_ + h * Dd + d;
    qn[o]   = f2bf(a * q0 + b * q1);
    kn[o]   = f2bf(a * k0 + b * k1);
    vn[o]   = f2bf(v);
    qnip[o] = f2bf(qh);
  }
}

// ---------------------------------------------------------------------------
// 6. Flash attention (main, roped q/k), 1 block = (head, 64-row q tile)
//    128 threads = 4 waves, each wave owns 16 q rows.
// ---------------------------------------------------------------------------
__global__ __launch_bounds__(128) void k_flash_attn(
    const unsigned short* __restrict__ qn, const unsigned short* __restrict__ kn,
    const unsigned short* __restrict__ vn, float* __restrict__ attn1) {
  __shared__ unsigned short Qs[64 * 136];
  __shared__ unsigned short Ks[64 * 136];
  __shared__ unsigned short Vs[128 * 72];     // transposed: [d][key]
  __shared__ unsigned short Ps[4 * 16 * 72];  // per-wave P tile (16x64)
  int h  = blockIdx.y;
  int q0 = blockIdx.x * 64;
  int tid = threadIdx.x, lane = tid & 31, wave = tid >> 5;

  // stage Q tile 64x128 (LDS row stride 136 halfs: pad 4 DWORDs / 64 DWORDs)
#if HAVE_TDM
  if (wave == 0) {
    tdm_load_tile_2d(ldsOffset(Qs), qn + (size_t)q0 * HID_ + h * Dd,
                     128, 64, 128, 64, HID_, /*interval(64 dw)=*/5, /*amount(4 dw)=*/3);
    __builtin_amdgcn_s_wait_tensorcnt(0);
  }
#else
  {
    int r = tid >> 1, ch = (tid & 1) * 64;
    const ushort8* gp =
        reinterpret_cast<const ushort8*>(qn + (size_t)(q0 + r) * HID_ + h * Dd + ch);
    ushort8* sp = reinterpret_cast<ushort8*>(Qs + r * 136 + ch);
    #pragma unroll
    for (int i = 0; i < 8; i++) sp[i] = gp[i];
  }
#endif
  __syncthreads();

  v16bf qf[4];
  #pragma unroll
  for (int kc = 0; kc < 4; kc++) qf[kc] = ldsFragA(Qs, 136, wave * 16, kc * 32);

  v8f o[8];
  #pragma unroll
  for (int di = 0; di < 8; di++)
    #pragma unroll
    for (int j = 0; j < 8; j++) o[di][j] = 0.f;
  float mrow[8], lsum[8];
  #pragma unroll
  for (int j = 0; j < 8; j++) { mrow[j] = -3e38f; lsum[j] = 0.f; }
  const float sc = 0.08838834764831845f;  // 1/sqrt(128)

  for (int t = 0; t < Ll / 64; t++) {
    int kbase = t * 64;
    __syncthreads();  // previous PV reads of Ks/Vs done
    // stage K tile 64x128
#if HAVE_TDM
    if (wave == 0)
      tdm_load_tile_2d(ldsOffset(Ks), kn + (size_t)kbase * HID_ + h * Dd,
                       128, 64, 128, 64, HID_, 5, 3);
#else
    {
      int r = tid >> 1, ch = (tid & 1) * 64;
      const ushort8* gp =
          reinterpret_cast<const ushort8*>(kn + (size_t)(kbase + r) * HID_ + h * Dd + ch);
      ushort8* sp = reinterpret_cast<ushort8*>(Ks + r * 136 + ch);
      #pragma unroll
      for (int i = 0; i < 8; i++) sp[i] = gp[i];
    }
#endif
    {  // stage V transposed 128x64
      int key = tid & 63, dh = (tid >> 6) * 64;
      const ushort8* gp =
          reinterpret_cast<const ushort8*>(vn + (size_t)(kbase + key) * HID_ + h * Dd + dh);
      #pragma unroll
      for (int i = 0; i < 8; i++) {
        ushort8 u = gp[i];
        #pragma unroll
        for (int e = 0; e < 8; e++) Vs[(dh + i * 8 + e) * 72 + key] = u[e];
      }
    }
#if HAVE_TDM
    if (wave == 0) __builtin_amdgcn_s_wait_tensorcnt(0);
#endif
    __syncthreads();

    // S = Q * K^T (16x64 per wave)
    v8f sAcc[4];
    #pragma unroll
    for (int ni = 0; ni < 4; ni++) {
      #pragma unroll
      for (int j = 0; j < 8; j++) sAcc[ni][j] = 0.f;
      #pragma unroll
      for (int kc = 0; kc < 4; kc++) {
        v16bf kf = ldsFragB(Ks, 136, ni * 16, kc * 32);
        sAcc[ni] = wmma_bf16(qf[kc], kf, sAcc[ni]);
      }
    }
    // online softmax
    float mcur[8];
    #pragma unroll
    for (int j = 0; j < 8; j++) mcur[j] = -3e38f;
    #pragma unroll
    for (int ni = 0; ni < 4; ni++)
      #pragma unroll
      for (int j = 0; j < 8; j++) mcur[j] = fmaxf(mcur[j], sAcc[ni][j]);
    #pragma unroll
    for (int msk = 1; msk < 16; msk <<= 1)
      #pragma unroll
      for (int j = 0; j < 8; j++) mcur[j] = fmaxf(mcur[j], __shfl_xor(mcur[j], msk, 32));
    float corr[8];
    #pragma unroll
    for (int j = 0; j < 8; j++) {
      float mnew = fmaxf(mrow[j], mcur[j] * sc);
      corr[j] = __expf(mrow[j] - mnew);
      mrow[j] = mnew;
      lsum[j] *= corr[j];
    }
    #pragma unroll
    for (int ni = 0; ni < 4; ni++)
      #pragma unroll
      for (int j = 0; j < 8; j++) {
        float p = __expf(sAcc[ni][j] * sc - mrow[j]);
        lsum[j] += p;
        int mr = ((lane >> 4) << 3) + j;
        Ps[(wave * 16 + mr) * 72 + ni * 16 + (lane & 15)] = f2bf(p);
      }
    #pragma unroll
    for (int di = 0; di < 8; di++)
      #pragma unroll
      for (int j = 0; j < 8; j++) o[di][j] *= corr[j];
    __syncthreads();

    // O += P * V
    v16bf pa[2];
    pa[0] = ldsFragA(Ps + wave * 16 * 72, 72, 0, 0);
    pa[1] = ldsFragA(Ps + wave * 16 * 72, 72, 0, 32);
    #pragma unroll
    for (int di = 0; di < 8; di++)
      #pragma unroll
      for (int kc = 0; kc < 2; kc++) {
        v16bf vf = ldsFragB(Vs, 72, di * 16, kc * 32);
        o[di] = wmma_bf16(pa[kc], vf, o[di]);
      }
  }
  // normalize and store (attn1 layout [L][H*D] fp32)
  #pragma unroll
  for (int msk = 1; msk < 16; msk <<= 1)
    #pragma unroll
    for (int j = 0; j < 8; j++) lsum[j] += __shfl_xor(lsum[j], msk, 32);
  #pragma unroll
  for (int di = 0; di < 8; di++)
    #pragma unroll
    for (int j = 0; j < 8; j++) {
      int m = ((lane >> 4) << 3) + j;
      int l = q0 + wave * 16 + m;
      int d = di * 16 + (lane & 15);
      attn1[(size_t)l * HID_ + h * Dd + d] = o[di][j] / lsum[j];
    }
}

// ---------------------------------------------------------------------------
// 7. IP attention (64 keys, single tile) + combine into fused[:, :3072] (bf16)
// ---------------------------------------------------------------------------
__global__ __launch_bounds__(128) void k_ip_attn(
    const unsigned short* __restrict__ qnip, const unsigned short* __restrict__ ipk,
    const unsigned short* __restrict__ ipv, const float* __restrict__ attn1,
    const float* __restrict__ ipScalePtr, unsigned short* __restrict__ fused) {
  __shared__ unsigned short Qs[64 * 136];
  __shared__ unsigned short Ks[64 * 136];
  __shared__ unsigned short Vs[128 * 72];
  __shared__ unsigned short Ps[4 * 16 * 72];
  int h  = blockIdx.y;
  int q0 = blockIdx.x * 64;
  int tid = threadIdx.x, lane = tid & 31, wave = tid >> 5;

#if HAVE_TDM
  if (wave == 0) {
    tdm_load_tile_2d(ldsOffset(Qs), qnip + (size_t)q0 * HID_ + h * Dd,
                     128, 64, 128, 64, HID_, 5, 3);
    tdm_load_tile_2d(ldsOffset(Ks), ipk + h * Dd,
                     128, 64, 128, 64, HID_, 5, 3);
  }
#else
  {  // Q
    int r = tid >> 1, ch = (tid & 1) * 64;
    const ushort8* gp =
        reinterpret_cast<const ushort8*>(qnip + (size_t)(q0 + r) * HID_ + h * Dd + ch);
    ushort8* sp = reinterpret_cast<ushort8*>(Qs + r * 136 + ch);
    #pragma unroll
    for (int i = 0; i < 8; i++) sp[i] = gp[i];
  }
  {  // K: ipk is [LIP][H*D]
    int r = tid >> 1, ch = (tid & 1) * 64;
    const ushort8* gp =
        reinterpret_cast<const ushort8*>(ipk + (size_t)r * HID_ + h * Dd + ch);
    ushort8* sp = reinterpret_cast<ushort8*>(Ks + r * 136 + ch);
    #pragma unroll
    for (int i = 0; i < 8; i++) sp[i] = gp[i];
  }
#endif
  {  // V transposed
    int key = tid & 63, dh = (tid >> 6) * 64;
    const ushort8* gp =
        reinterpret_cast<const ushort8*>(ipv + (size_t)key * HID_ + h * Dd + dh);
    #pragma unroll
    for (int i = 0; i < 8; i++) {
      ushort8 u = gp[i];
      #pragma unroll
      for (int e = 0; e < 8; e++) Vs[(dh + i * 8 + e) * 72 + key] = u[e];
    }
  }
#if HAVE_TDM
  if (wave == 0) __builtin_amdgcn_s_wait_tensorcnt(0);
#endif
  __syncthreads();

  v16bf qf[4];
  #pragma unroll
  for (int kc = 0; kc < 4; kc++) qf[kc] = ldsFragA(Qs, 136, wave * 16, kc * 32);

  v8f sAcc[4];
  #pragma unroll
  for (int ni = 0; ni < 4; ni++) {
    #pragma unroll
    for (int j = 0; j < 8; j++) sAcc[ni][j] = 0.f;
    #pragma unroll
    for (int kc = 0; kc < 4; kc++) {
      v16bf kf = ldsFragB(Ks, 136, ni * 16, kc * 32);
      sAcc[ni] = wmma_bf16(qf[kc], kf, sAcc[ni]);
    }
  }
  const float sc = 0.08838834764831845f;
  float mcur[8], lsum[8];
  #pragma unroll
  for (int j = 0; j < 8; j++) { mcur[j] = -3e38f; lsum[j] = 0.f; }
  #pragma unroll
  for (int ni = 0; ni < 4; ni++)
    #pragma unroll
    for (int j = 0; j < 8; j++) mcur[j] = fmaxf(mcur[j], sAcc[ni][j]);
  #pragma unroll
  for (int msk = 1; msk < 16; msk <<= 1)
    #pragma unroll
    for (int j = 0; j < 8; j++) mcur[j] = fmaxf(mcur[j], __shfl_xor(mcur[j], msk, 32));
  #pragma unroll
  for (int ni = 0; ni < 4; ni++)
    #pragma unroll
    for (int j = 0; j < 8; j++) {
      float p = __expf((sAcc[ni][j] - mcur[j]) * sc);
      lsum[j] += p;
      int mr = ((lane >> 4) << 3) + j;
      Ps[(wave * 16 + mr) * 72 + ni * 16 + (lane & 15)] = f2bf(p);
    }
  __syncthreads();

  v8f o[8];
  #pragma unroll
  for (int di = 0; di < 8; di++)
    #pragma unroll
    for (int j = 0; j < 8; j++) o[di][j] = 0.f;
  v16bf pa[2];
  pa[0] = ldsFragA(Ps + wave * 16 * 72, 72, 0, 0);
  pa[1] = ldsFragA(Ps + wave * 16 * 72, 72, 0, 32);
  #pragma unroll
  for (int di = 0; di < 8; di++)
    #pragma unroll
    for (int kc = 0; kc < 2; kc++) {
      v16bf vf = ldsFragB(Vs, 72, di * 16, kc * 32);
      o[di] = wmma_bf16(pa[kc], vf, o[di]);
    }
  #pragma unroll
  for (int msk = 1; msk < 16; msk <<= 1)
    #pragma unroll
    for (int j = 0; j < 8; j++) lsum[j] += __shfl_xor(lsum[j], msk, 32);

  float ipsc = *ipScalePtr;
  #pragma unroll
  for (int di = 0; di < 8; di++)
    #pragma unroll
    for (int j = 0; j < 8; j++) {
      int m = ((lane >> 4) << 3) + j;
      int l = q0 + wave * 16 + m;
      int d = di * 16 + (lane & 15);
      float a1 = attn1[(size_t)l * HID_ + h * Dd + d];
      float v  = o[di][j] / lsum[j];
      fused[(size_t)l * FUSED_ + h * Dd + d] = f2bf(a1 + ipsc * v);
    }
}

// ---------------------------------------------------------------------------
// Launch
// ---------------------------------------------------------------------------
extern "C" void kernel_launch(void* const* d_in, const int* in_sizes, int n_in,
                              void* d_out, int out_size, void* d_ws, size_t ws_size,
                              hipStream_t stream) {
  const float* x        = (const float*)d_in[0];
  const float* vec      = (const float*)d_in[1];
  const float* pe       = (const float*)d_in[2];
  const float* imgproj  = (const float*)d_in[3];
  const float* ip_scale = (const float*)d_in[4];
  const float* mod_w    = (const float*)d_in[5];
  const float* mod_b    = (const float*)d_in[6];
  const float* l1w      = (const float*)d_in[7];
  const float* l1b      = (const float*)d_in[8];
  const float* l2w      = (const float*)d_in[9];
  const float* l2b      = (const float*)d_in[10];
  const float* qnsc     = (const float*)d_in[11];
  const float* knsc     = (const float*)d_in[12];
  const float* ipkw     = (const float*)d_in[13];
  const float* ipvw     = (const float*)d_in[14];

  auto au = [](size_t v) { return (v + 255) & ~size_t(255); };
  char* p = (char*)d_ws;
  float*          mod   = (float*)p;          p += au((size_t)QKV_ * 4);
  unsigned short* xmod  = (unsigned short*)p; p += au((size_t)Ll * HID_ * 2);
  float*          qkv   = (float*)p;          p += au((size_t)Ll * QKV_ * 4);
  unsigned short* fused = (unsigned short*)p; p += au((size_t)Ll * FUSED_ * 2);
  unsigned short* qn    = (unsigned short*)p; p += au((size_t)Ll * HID_ * 2);
  unsigned short* kn    = (unsigned short*)p; p += au((size_t)Ll * HID_ * 2);
  unsigned short* vn    = (unsigned short*)p; p += au((size_t)Ll * HID_ * 2);
  unsigned short* qnip  = (unsigned short*)p; p += au((size_t)Ll * HID_ * 2);
  unsigned short* ipjb  = (unsigned short*)p; p += au((size_t)LIP_ * CTX_ * 2);
  unsigned short* ipk   = (unsigned short*)p; p += au((size_t)LIP_ * HID_ * 2);
  unsigned short* ipv   = (unsigned short*)p; p += au((size_t)LIP_ * HID_ * 2);
  float*          attn1 = (float*)p;          p += au((size_t)Ll * HID_ * 4);

  // 1. modulation vector
  k_mod_gemv<<<QKV_ / 8, 256, 0, stream>>>(vec, mod_w, mod_b, mod);
  // 2. layernorm + modulate -> bf16
  k_ln_modulate<<<Ll, 256, 0, stream>>>(x, mod, xmod);
  // 3a. linear1 (qkv part), fp32 out
  k_gemm<0><<<dim3(QKV_ / BN, Ll / BM), 256, 0, stream>>>(
      xmod, HID_, l1w, HID_, l1b, qkv, QKV_, Ll, QKV_, HID_, nullptr, nullptr);
  // 3b. linear1 (mlp part), gelu -> bf16 into fused[:, 3072:]
  k_gemm<1><<<dim3(MLP_ / BN, Ll / BM), 256, 0, stream>>>(
      xmod, HID_, l1w + (size_t)QKV_ * HID_, HID_, l1b + QKV_,
      fused + HID_, FUSED_, Ll, MLP_, HID_, nullptr, nullptr);
  // 4. image_proj -> bf16, ip_k / ip_v projections (bf16 out)
  k_f32_to_bf16<<<(LIP_ * CTX_ + 255) / 256, 256, 0, stream>>>(imgproj, ipjb, LIP_ * CTX_);
  k_gemm<2><<<dim3(HID_ / BN, 1), 256, 0, stream>>>(
      ipjb, CTX_, ipkw, CTX_, nullptr, ipk, HID_, LIP_, HID_, CTX_, nullptr, nullptr);
  k_gemm<2><<<dim3(HID_ / BN, 1), 256, 0, stream>>>(
      ipjb, CTX_, ipvw, CTX_, nullptr, ipv, HID_, LIP_, HID_, CTX_, nullptr, nullptr);
  // 5. q/k rmsnorm + rope -> bf16
  k_qkv_postproc<<<Ll, 128, 0, stream>>>(qkv, pe, qnsc, knsc, qn, kn, vn, qnip);
  // 6. self attention
  k_flash_attn<<<dim3(Ll / 64, Hh), 128, 0, stream>>>(qn, kn, vn, attn1);
  // 7. ip attention + combine -> fused[:, :3072]
  k_ip_attn<<<dim3(Ll / 64, Hh), 128, 0, stream>>>(qnip, ipk, ipv, attn1, ip_scale, fused);
  // 8. linear2 + residual gate -> d_out
  k_gemm<3><<<dim3(HID_ / BN, Ll / BM), 256, 0, stream>>>(
      fused, FUSED_, l2w, FUSED_, l2b, d_out, HID_, Ll, HID_, FUSED_,
      x, mod + 2 * HID_);
}
